// RNNFC_32856499814962
// MI455X (gfx1250) — matrix operand order repaired and checked by reading, not compiled, not run on hardware
//
#include <hip/hip_runtime.h>
#include <hip/hip_bf16.h>
#include <math.h>

// ---------------------------------------------------------------------------
// RNN: xp = x@W_ih + b_h   (parallel GEMM, kernel 1; xp stored in WMMA C-frag
//                           layout keyed to the recurrence's tiles)
//      h_t = tanh(xp_t + h_{t-1}@W_hh) * silence   (sequential, kernel 2)
//      out = hs@W_out + b_out  (parallel GEMM, kernel 3)
// All matmuls: v_wmma_f32_16x16x32_bf16 (fp32 accumulate, bf16 operands).
// All LDS fragment accesses use one per-lane base pointer + compile-time
// constant offsets so they lower to ds_load_b128 with immediate offsets
// (no per-fragment address VALU -> no WMMA->VALU hazard NOPs).
// ---------------------------------------------------------------------------

typedef __attribute__((ext_vector_type(16))) __bf16 v16bf;
typedef __attribute__((ext_vector_type(8)))  __bf16 v8bf;
typedef __attribute__((ext_vector_type(8)))  float  v8f;

#define Bsz 64
#define Tsz 2048
#define INsz 128
#define HIDsz 256
#define OUTsz 128

__device__ __forceinline__ v8f wmma_bf16(v16bf a, v16bf b, v8f c) {
  return __builtin_amdgcn_wmma_f32_16x16x32_bf16(false, a, false, b, (short)0, c, false, false);
}

__device__ __forceinline__ v8f zero8() { return (v8f){0,0,0,0,0,0,0,0}; }

__device__ __forceinline__ v16bf combine16(v8bf lo, v8bf hi) {
  return __builtin_shufflevector(lo, hi, 0,1,2,3,4,5,6,7,8,9,10,11,12,13,14,15);
}

__device__ __forceinline__ float fast_tanh(float x) {
#if __has_builtin(__builtin_amdgcn_tanhf)
  return __builtin_amdgcn_tanhf(x);   // v_tanh_f32 (gfx1250 TRANS op)
#else
  return tanhf(x);
#endif
}

// A-fragment (16x32 bf16) from a row-major fp32 row pointer.
// lane l: M = l&15, kbase=(l>>4)*8; elem j<8 -> K=kb+kbase+j ; j>=8 -> +16.
__device__ __forceinline__ v16bf gather_A_f32(const float* __restrict__ rowbase,
                                              int kb, int lane) {
  const int kbase = (lane >> 4) << 3;
  const float* p0 = rowbase + kb + kbase;
  v16bf a;
#pragma unroll
  for (int j = 0; j < 8; ++j) {
    a[j]     = (__bf16)p0[j];
    a[8 + j] = (__bf16)p0[16 + j];
  }
  return a;
}

// Same fragment from a row-major bf16 row: two 16-byte loads at const offsets.
__device__ __forceinline__ v16bf gather_A_bf16(const __bf16* __restrict__ rowbase,
                                               int kb) {
  v8bf lo = *(const v8bf*)(rowbase + kb);
  v8bf hi = *(const v8bf*)(rowbase + kb + 16);
  return combine16(lo, hi);
}

// B-fragment (32x16 bf16) gathered from row-major fp32 W[K][ldn].
// lane l: N = ncol+(l&15), elem j -> K = kb + (l>>4)*8 + j (+8 for j>=8).
__device__ __forceinline__ v16bf gather_B_f32(const float* __restrict__ W, int ldn,
                                              int kb, int ncol, int lane) {
  const int khalf = (lane >> 4) << 3;
  const float* p = W + (long)(kb + khalf) * ldn + ncol + (lane & 15);
  v16bf b;
#pragma unroll
  for (int j = 0; j < 16; ++j) b[j] = (__bf16)p[(long)(j + (j & 8)) * ldn];
  return b;
}

// Pre-swizzle fp32 weights into LDS as bf16 B-fragments:
// Bs[frag][lane][16], frag = kb8*nnt + nt (32 bytes per lane per fragment).
__device__ __forceinline__ void stage_B(const float* __restrict__ W, int ldn,
                                        int nkb, int nnt, __bf16* Bs) {
  const int lane = threadIdx.x & 31, wave = threadIdx.x >> 5;
  const int nf = nkb * nnt;
  for (int f = wave; f < nf; f += 8) {
    const int kb8 = f / nnt, nt = f - kb8 * nnt;
    const int khalf = (lane >> 4) << 3;
    const float* p = W + (long)(kb8 * 32 + khalf) * ldn + nt * 16 + (lane & 15);
    __bf16* d = Bs + ((long)f * 32 + lane) * 16;
#pragma unroll
    for (int j = 0; j < 16; ++j) d[j] = (__bf16)p[(long)(j + (j & 8)) * ldn];
  }
}

// Fragment load from a precomputed per-lane base + constant element offset.
__device__ __forceinline__ v16bf load_frag(const __bf16* base, int elemOfs) {
  return *(const v16bf*)(base + elemOfs);
}

// ---------------------------------------------------------------------------
// Kernel 1: xp = x@W_ih + b_h, M ordered as [t][b] so M-tile (t*4+btile)
// matches the recurrence's batch tile at time t. Output stored in C-fragment
// layout: xp[((mtile*16 + ntile)*32 + lane)*8 + r]  (one v8f store per tile).
// ---------------------------------------------------------------------------
__global__ __launch_bounds__(256) void k_inproj(const float* __restrict__ x,
                                                const float* __restrict__ W_ih,
                                                const float* __restrict__ b_h,
                                                float* __restrict__ xp) {
  __shared__ __align__(32) __bf16 Bs[4 * 16 * 32 * 16];  // 64 KB
  stage_B(W_ih, HIDsz, 4, 16, Bs);
  __syncthreads();

  const int lane = threadIdx.x & 31;
  const int wave = threadIdx.x >> 5;
  const int mtile = blockIdx.x * 4 + (wave >> 1);   // 8192 M-tiles, [t][b] order
  const int nth = (wave & 1) * 8;                   // N-tile base (0 or 8)
  const int m = lane & 15;

  // per-lane B base with nth folded in: frag (kb8, nth+nt) sits at
  // Bsb + kb8*8192 + nt*512 (elements) -> constant ds offsets <= 56 KB
  const __bf16* Bsb = Bs + nth * 512 + lane * 16;

  // M row = mtile*16 + m  ->  t = mtile>>2, b = (mtile&3)*16 + m
  const int tt = mtile >> 2;
  const int bb = (mtile & 3) * 16 + m;
  const float* xrow = x + ((long)bb * Tsz + tt) * INsz;

  v16bf af[4];
#pragma unroll
  for (int kb8 = 0; kb8 < 4; ++kb8) af[kb8] = gather_A_f32(xrow, kb8 * 32, lane);

  v8f acc[8];
#pragma unroll
  for (int nt = 0; nt < 8; ++nt) acc[nt] = zero8();

#pragma unroll
  for (int kb8 = 0; kb8 < 4; ++kb8) {
    v16bf bf[8];
#pragma unroll
    for (int nt = 0; nt < 8; ++nt) bf[nt] = load_frag(Bsb, kb8 * 8192 + nt * 512);
#pragma unroll
    for (int nt = 0; nt < 8; ++nt) acc[nt] = wmma_bf16(af[kb8], bf[nt], acc[nt]);
  }

#pragma unroll
  for (int nt = 0; nt < 8; ++nt) {
    const float bias = b_h[(nth + nt) * 16 + m];
    float* q = xp + (((long)mtile * 16 + nth + nt) * 32 + lane) * 8;
    *(v8f*)q = acc[nt] + bias;  // fragment-layout store, fully coalesced
  }
}

// ---------------------------------------------------------------------------
// Kernel 2: recurrence. 4 workgroups (16 batch rows each), 8 waves.
// Wave w owns N-tiles {2w,2w+1}; W_hh lives in registers (16 B-fragments).
// h double-buffered in LDS; 4 independent WMMA chains; xp read as 2 coalesced
// v8f fragment loads + register prefetch of t+1 + global_prefetch of t+2;
// hs written via coalesced LDS->global copy-out of the previous step's h.
// ---------------------------------------------------------------------------
#define HSTRIDE 264  // padded LDS row stride (elements)

__global__ __launch_bounds__(256) void k_recur(const float* __restrict__ xp,
                                               const float* __restrict__ W_hh,
                                               const float* __restrict__ silence,
                                               __bf16* __restrict__ hs) {
  __shared__ __align__(32) __bf16 hbuf[2][16][HSTRIDE];

  const int lane = threadIdx.x & 31;
  const int wave = threadIdx.x >> 5;
  const int btile = blockIdx.x;  // 4 batch tiles
  const int m = lane & 15, hi = lane >> 4;

  for (int i = threadIdx.x; i < 2 * 16 * HSTRIDE; i += 256)
    (&hbuf[0][0][0])[i] = (__bf16)0.0f;  // h_{-1} = 0

  // W_hh B-fragments -> registers (one-time, L2-hot)
  v16bf Bf[2][8];
#pragma unroll
  for (int nt2 = 0; nt2 < 2; ++nt2) {
    const int ncol = (wave * 2 + nt2) * 16;
#pragma unroll
    for (int kb8 = 0; kb8 < 8; ++kb8)
      Bf[nt2][kb8] = gather_B_f32(W_hh, HIDsz, kb8 * 32, ncol, lane);
  }
  const float sd0 = silence[(wave * 2 + 0) * 16 + m];
  const float sd1 = silence[(wave * 2 + 1) * 16 + m];

  // xp fragment pointers: tile id at step t = (t*4 + btile)*16 + ntile
  const float* xpw = xp + (((long)btile * 16 + wave * 2) * 32 + lane) * 8;
  const long tstep = (long)64 * 32 * 8;  // 4 btiles * 16 ntiles * 32 lanes * 8
  const float* pf0 = xpw;                 // covers tile nt0's full 1KB
  const float* pf1 = xpw + 32 * 8;        // covers tile nt1's full 1KB

  // per-lane bases for A-fragment reads and h writes (const offsets inside)
  const int kbase = hi * 8;
  const __bf16* hrd0 = &hbuf[0][m][kbase];          // + kb (+16) const
  const __bf16* hrd1 = &hbuf[1][m][kbase];
  __bf16* hwr0 = &hbuf[0][8 * hi][wave * 32 + m];   // + r*HSTRIDE (+16) const
  __bf16* hwr1 = &hbuf[1][8 * hi][wave * 32 + m];

  // coalesced copy-out mapping: thread -> (row, 16 cols) of h
  const int crow = threadIdx.x >> 4;
  const int ccol = (threadIdx.x & 15) * 16;
  const __bf16* crd0 = &hbuf[0][crow][ccol];
  const __bf16* crd1 = &hbuf[1][crow][ccol];
  __bf16* hsout = hs + ((long)(btile * 16 + crow) * Tsz) * HIDsz + ccol;

  // register-prefetch xp for t=0
  v8f xpr0 = *(const v8f*)(xpw);
  v8f xpr1 = *(const v8f*)(xpw + 32 * 8);

  __syncthreads();

  for (int t = 0; t < Tsz; ++t) {
    const int rb = (t + 1) & 1, wb = t & 1;
    const __bf16* hrd = rb ? hrd1 : hrd0;
    __bf16*       hwr = wb ? hwr1 : hwr0;

    // stream h_{t-1} (complete in read buffer) to global hs, coalesced
    if (t > 0) {
      const __bf16* crd = rb ? crd1 : crd0;
      v8bf c0 = *(const v8bf*)(crd);
      v8bf c1 = *(const v8bf*)(crd + 8);
      *(v16bf*)(hsout + (long)(t - 1) * HIDsz) = combine16(c0, c1);
    }

    // h_{t-1} @ W_hh : 4 independent accumulation chains
    v8f a0e = zero8(), a0o = zero8(), a1e = zero8(), a1o = zero8();
#pragma unroll
    for (int g = 0; g < 2; ++g) {
      v16bf af[4];
#pragma unroll
      for (int i = 0; i < 4; ++i) {
        const int kb = (g * 4 + i) * 32;
        af[i] = combine16(*(const v8bf*)(hrd + kb), *(const v8bf*)(hrd + kb + 16));
      }
#pragma unroll
      for (int i = 0; i < 4; ++i) {
        const int kb8 = g * 4 + i;
        if (i & 1) { a0o = wmma_bf16(af[i], Bf[0][kb8], a0o);
                     a1o = wmma_bf16(af[i], Bf[1][kb8], a1o); }
        else       { a0e = wmma_bf16(af[i], Bf[0][kb8], a0e);
                     a1e = wmma_bf16(af[i], Bf[1][kb8], a1e); }
      }
    }
    v8f s0 = a0e + a0o + xpr0;
    v8f s1 = a1e + a1o + xpr1;

    // h_t = tanh(.) * silence -> LDS write buffer (const offsets off hwr)
#pragma unroll
    for (int r = 0; r < 8; ++r) {
      hwr[r * HSTRIDE]      = (__bf16)(fast_tanh(s0[r]) * sd0);
      hwr[r * HSTRIDE + 16] = (__bf16)(fast_tanh(s1[r]) * sd1);
    }

    // register-prefetch xp[t+1]; hint-prefetch xp[t+2]
    if (t + 1 < Tsz) {
      const float* p = xpw + (long)(t + 1) * tstep;
      xpr0 = *(const v8f*)(p);
      xpr1 = *(const v8f*)(p + 32 * 8);
    }
    if (t + 2 < Tsz) {
      __builtin_prefetch(pf0 + (long)(t + 2) * tstep, 0, 1);
      __builtin_prefetch(pf1 + (long)(t + 2) * tstep, 0, 1);
    }
    __syncthreads();
  }

  // final h_{T-1} lives in hbuf[1]
  {
    v8bf c0 = *(const v8bf*)(crd1);
    v8bf c1 = *(const v8bf*)(crd1 + 8);
    *(v16bf*)(hsout + (long)(Tsz - 1) * HIDsz) = combine16(c0, c1);
  }
}

// ---------------------------------------------------------------------------
// Kernel 3: out[B*T,128] = hs[B*T,256] @ W_out[256,128] + b_out
// 8 waves, one 16-row M-tile each, 8 N-tiles; B frags batch-preloaded per kb.
// ---------------------------------------------------------------------------
__global__ __launch_bounds__(256) void k_outproj(const __bf16* __restrict__ hs,
                                                 const float* __restrict__ W_out,
                                                 const float* __restrict__ b_out,
                                                 float* __restrict__ out) {
  __shared__ __align__(32) __bf16 Bs[8 * 8 * 32 * 16];  // 64 KB
  stage_B(W_out, OUTsz, 8, 8, Bs);
  __syncthreads();

  const int lane = threadIdx.x & 31;
  const int wave = threadIdx.x >> 5;
  const int mtile = blockIdx.x * 8 + wave;
  const int m = lane & 15, hi = lane >> 4;
  // per-lane bases: A row with kbase folded, B with lane folded
  const __bf16* hrow = hs + ((long)mtile * 16 + m) * HIDsz + ((lane >> 4) << 3);
  const __bf16* Bsb  = Bs + lane * 16;  // frag (kb8,nt) at + kb8*4096 + nt*512

  v8f acc[8];
#pragma unroll
  for (int nt = 0; nt < 8; ++nt) acc[nt] = zero8();

#pragma unroll
  for (int kb8 = 0; kb8 < 8; ++kb8) {
    v16bf a = gather_A_bf16(hrow, kb8 * 32);
    v16bf bf[8];
#pragma unroll
    for (int nt = 0; nt < 8; ++nt) bf[nt] = load_frag(Bsb, kb8 * 4096 + nt * 512);
#pragma unroll
    for (int nt = 0; nt < 8; ++nt) acc[nt] = wmma_bf16(a, bf[nt], acc[nt]);
  }

#pragma unroll
  for (int nt = 0; nt < 8; ++nt) {
    const float bias = b_out[nt * 16 + m];
#pragma unroll
    for (int r = 0; r < 8; ++r) {
      const long orow = (long)mtile * 16 + r + 8 * hi;
      out[orow * OUTsz + nt * 16 + m] = acc[nt][r] + bias;
    }
  }
}

// ---------------------------------------------------------------------------
extern "C" void kernel_launch(void* const* d_in, const int* in_sizes, int n_in,
                              void* d_out, int out_size, void* d_ws, size_t ws_size,
                              hipStream_t stream) {
  const float* x       = (const float*)d_in[0];
  const float* W_ih    = (const float*)d_in[1];
  const float* W_hh    = (const float*)d_in[2];
  const float* b_h     = (const float*)d_in[3];
  const float* W_out   = (const float*)d_in[4];
  const float* b_out   = (const float*)d_in[5];
  const float* silence = (const float*)d_in[6];
  float* out = (float*)d_out;

  // workspace: xp fp32 (fragment layout) [B*T*HID], then hs bf16 [B*T*HID]
  char* ws = (char*)d_ws;
  float*  xp = (float*)ws;
  __bf16* hs = (__bf16*)(ws + (size_t)Bsz * Tsz * HIDsz * sizeof(float));

  const int mtiles = (Bsz * Tsz) / 16;  // 8192

  k_inproj<<<mtiles / 4, 256, 0, stream>>>(x, W_ih, b_h, xp);   // 2048 WGs
  k_recur<<<Bsz / 16, 256, 0, stream>>>(xp, W_hh, silence, hs); // 4 WGs
  k_outproj<<<mtiles / 8, 256, 0, stream>>>(hs, W_out, b_out, out); // 1024 WGs
}
